// ODEModel_32667521254061
// MI455X (gfx1250) — matrix operand order repaired and checked by reading, not compiled, not run on hardware
//
#include <hip/hip_runtime.h>
#include <hip/hip_bf16.h>

// ---------------------------------------------------------------------------
// Neural ODE (RK4, 40 steps) on gfx1250 via v_wmma_f32_16x16x32_bf16.
// 4 independent accumulator chains per wave, software-pipelined fragment
// loads enforced with sched_group_barrier; biases register-cached.
// ---------------------------------------------------------------------------

typedef __attribute__((ext_vector_type(16))) __bf16 vbf16;
typedef __attribute__((ext_vector_type(8)))  float  v8f;
typedef unsigned short u16;
typedef unsigned int   u32;

#define BATCH   16384
#define IN_DIM  512
#define HID     256
#define OUT_DIM 18
#define NSTEPS  40
#define DT      (1.0f / 40.0f)

// ---- workspace layout (bytes) ----
#define WS_H    0                          // f32 [16384][256]  = 16 MiB
#define WS_XBF  (16777216)                 // bf16 [16384][512] = 16 MiB
#define WS_WST  (WS_XBF + 16777216)        // bf16 [256][512]   (W_state^T)
#define WS_W1T  (WS_WST + 262144)          // bf16 [256][256]   (W_dyn1[:256]^T)
#define WS_W2T  (WS_W1T + 131072)          // bf16 [256][256]   (W_dyn2^T)

// ---- RK4 kernel LDS layout (byte offsets, padded strides) ----
#define WSTRIDE 528                         // 256 bf16 + 16B pad
#define SW1   0
#define SW2   (SW1 + 256 * WSTRIDE)         // 135168
#define SY    (SW2 + 256 * WSTRIDE)         // 270336
#define YSTR  528
#define SZ    (SY + 16 * YSTR)              // 278784
#define SH    (SZ + 16 * YSTR)              // 287232
#define HSTR  1040                          // 256 f32 + 16B pad
#define SG    (SH + 16 * HSTR)              // 303872
#define SMEM_RK4 (SG + 16 * HSTR)           // 320512  (< 327680)

#define H0_WSTR  1040                       // 512 bf16 + 16B pad
#define SMEM_H0  (256 * H0_WSTR)            // 266240

// sched_group_barrier masks (LLVM AMDGPU IGroupLP)
#define SGB_MFMA     0x008                  // MFMA / WMMA
#define SGB_VMEMRD   0x020
#define SGB_DSRD     0x100
#if defined(__has_builtin)
#if __has_builtin(__builtin_amdgcn_sched_group_barrier)
#define SGB(mask, n, id) __builtin_amdgcn_sched_group_barrier((mask), (n), (id))
#endif
#endif
#ifndef SGB
#define SGB(mask, n, id)
#endif

// ---------------------------------------------------------------------------
__device__ __forceinline__ u16 f2bf(float f) {
    u32 u = __float_as_uint(f);
    u += 0x7FFFu + ((u >> 16) & 1u);        // round-to-nearest-even
    return (u16)(u >> 16);
}

__device__ __forceinline__ v8f splat8(float x) {
    v8f v = {x, x, x, x, x, x, x, x};
    return v;
}

union FragU { uint4 q[2]; vbf16 v; };

// A-matrix fragment (16x32 bf16): lane m=L&15, half=L>>4.
// VGPR0..3: K = k0 + half*8 + {0..7}; VGPR4..7: K = k0 + 16 + half*8 + {0..7}
__device__ __forceinline__ vbf16 loadA(const char* buf, int stride, int m,
                                       int hl, int k0) {
    FragU f;
    const char* p = buf + m * stride + (k0 + hl * 8) * 2;
    f.q[0] = *(const uint4*)(p);
    f.q[1] = *(const uint4*)(p + 32);       // +16 K values * 2B
    return f.v;
}

// B-matrix fragment (32x16 bf16), stored transposed [n][k]: lane n=L&15,
// VGPR0..3: K = k0 + half*8 + {0..7}; VGPR4..7: K = k0 + 16 + half*8 + {0..7}
__device__ __forceinline__ vbf16 loadB(const char* buf, int stride, int n,
                                       int hl, int k0) {
    FragU f;
    const char* p = buf + n * stride + (k0 + hl * 8) * 2;
    f.q[0] = *(const uint4*)(p);
    f.q[1] = *(const uint4*)(p + 32);
    return f.v;
}

__device__ __forceinline__ v8f wmma_bf16(vbf16 a, vbf16 b, v8f c) {
    return __builtin_amdgcn_wmma_f32_16x16x32_bf16(
        false, a, false, b, (short)0, c, false, false);
}

// 4-chain GEMM over KSTEPS k-steps, software-pipelined: fragments for step
// k+1 are loaded into fresh registers before the 4 WMMAs of step k issue.
// sched_group_barrier pins the (loads; 4xWMMA) issue pattern so the
// register allocator cannot collapse the double buffer.
template <int KSTEPS, bool AGLB>
__device__ __forceinline__ void gemm4(const char* abuf, int astr,
                                      const char* bbuf, int bstr, int m,
                                      int hl, int ncb, v8f acc[4]) {
    vbf16 a_cur = loadA(abuf, astr, m, hl, 0);
    vbf16 b_cur[4];
#pragma unroll
    for (int t = 0; t < 4; t++) b_cur[t] = loadB(bbuf, bstr, ncb + t * 16, hl, 0);
#pragma unroll
    for (int k = 0; k < KSTEPS; k++) {
        vbf16 a_nxt;
        vbf16 b_nxt[4];
        if (k < KSTEPS - 1) {
            a_nxt = loadA(abuf, astr, m, hl, (k + 1) * 32);
#pragma unroll
            for (int t = 0; t < 4; t++)
                b_nxt[t] = loadB(bbuf, bstr, ncb + t * 16, hl, (k + 1) * 32);
        }
#pragma unroll
        for (int t = 0; t < 4; t++) acc[t] = wmma_bf16(a_cur, b_cur[t], acc[t]);
        // issue pattern per iteration: next-step loads, then 4 WMMAs
        if (AGLB) {
            SGB(SGB_VMEMRD, 2, 0);
            SGB(SGB_DSRD, 8, 0);
        } else {
            SGB(SGB_DSRD, 10, 0);
        }
        SGB(SGB_MFMA, 4, 0);
        if (k < KSTEPS - 1) {
            a_cur = a_nxt;
#pragma unroll
            for (int t = 0; t < 4; t++) b_cur[t] = b_nxt[t];
        }
    }
}

// ---------------------------------------------------------------------------
__global__ void cvt_x_kernel(const float* __restrict__ x, u16* __restrict__ xbf,
                             int n) {
    for (int i = blockIdx.x * blockDim.x + threadIdx.x; i < n;
         i += gridDim.x * blockDim.x)
        xbf[i] = f2bf(x[i]);
}

// blockIdx.x = n (0..255), threadIdx.x = k (0..511)
__global__ void prep_w_kernel(const float* __restrict__ W_state,
                              const float* __restrict__ W_dyn1,
                              const float* __restrict__ W_dyn2,
                              u16* __restrict__ Wst, u16* __restrict__ W1t,
                              u16* __restrict__ W2t) {
    int n = blockIdx.x, k = threadIdx.x;
    Wst[n * 512 + k] = f2bf(W_state[k * 256 + n]);
    if (k < 256) {
        W1t[n * 256 + k] = f2bf(W_dyn1[k * 256 + n]);
        W2t[n * 256 + k] = f2bf(W_dyn2[k * 256 + n]);
    }
}

// ---------------------------------------------------------------------------
// h0 = tanh(x @ W_state + b_state)   [16384,512]x[512,256]
// block: 128 threads (4 waves), 16 rows; W_state^T resident in LDS.
__global__ __launch_bounds__(128) void h0_kernel(
    const u16* __restrict__ xbf, const u16* __restrict__ Wst,
    const float* __restrict__ b_state, float* __restrict__ h) {
    extern __shared__ char sm[];
    const int tid = threadIdx.x;
    const int wave = tid >> 5, lane = tid & 31;
    const int hl = lane >> 4, l16 = lane & 15;
    const int ncb = wave * 64 + l16;          // chain t -> col ncb + t*16
    const long rowBase = (long)blockIdx.x * 16;

    float rb[4];
#pragma unroll
    for (int t = 0; t < 4; t++) rb[t] = b_state[ncb + t * 16];

    for (int r = tid; r < 256; r += 128) {
        const uint4* s = (const uint4*)(Wst + r * 512);
        uint4* d = (uint4*)(sm + r * H0_WSTR);
#pragma unroll
        for (int i = 0; i < 64; i++) d[i] = s[i];
    }
    __syncthreads();

    const char* xrow = (const char*)xbf + rowBase * IN_DIM * 2;
    v8f acc[4];
#pragma unroll
    for (int t = 0; t < 4; t++) acc[t] = splat8(rb[t]);
    gemm4<16, true>(xrow, IN_DIM * 2, sm, H0_WSTR, l16, hl, ncb, acc);
#pragma unroll
    for (int t = 0; t < 4; t++) {
        const int ncol = ncb + t * 16;
#pragma unroll
        for (int r = 0; r < 8; r++) {
            int m = r + hl * 8;
            h[(rowBase + m) * HID + ncol] = tanhf(acc[t][r]);
        }
    }
}

// ---------------------------------------------------------------------------
// One RK4 step. block: 128 threads (4 waves), owns 16 batch rows.
// dyn(y,t): z = tanh(y@W1 + t*w1row + b1); dh = z@W2 + b2
__global__ __launch_bounds__(128) void rk4_step_kernel(
    float* __restrict__ h, const u16* __restrict__ W1t,
    const u16* __restrict__ W2t, const float* __restrict__ W_dyn1,
    const float* __restrict__ b1, const float* __restrict__ b2, float t0) {
    extern __shared__ char sm[];
    const int tid = threadIdx.x;
    const int wave = tid >> 5, lane = tid & 31;
    const int hl = lane >> 4, l16 = lane & 15;
    const int ncb = wave * 64 + l16;          // chain t -> col ncb + t*16
    const long rowBase = (long)blockIdx.x * 16;
    const float* __restrict__ w1row = W_dyn1 + 256 * 256;  // time row

    // register-cache per-lane biases (loop-invariant across stages)
    float rb1[4], rb2[4], rwt[4];
#pragma unroll
    for (int t = 0; t < 4; t++) {
        const int ncol = ncb + t * 16;
        rb1[t] = b1[ncol];
        rb2[t] = b2[ncol];
        rwt[t] = w1row[ncol];
    }

    // stage weights into LDS (padded rows)
    for (int r = tid; r < 256; r += 128) {
        const uint4* s1 = (const uint4*)(W1t + r * 256);
        const uint4* s2 = (const uint4*)(W2t + r * 256);
        uint4* d1 = (uint4*)(sm + SW1 + r * WSTRIDE);
        uint4* d2 = (uint4*)(sm + SW2 + r * WSTRIDE);
#pragma unroll
        for (int i = 0; i < 32; i++) { d1[i] = s1[i]; d2[i] = s2[i]; }
    }
    // h tile -> sH (f32), sG (running RK4 combo), sY (bf16 stage input y1=h)
    for (int idx = tid; idx < 16 * 64; idx += 128) {
        int r = idx >> 6, c = idx & 63;  // c = uint4 index (4 floats)
        uint4 v = *(const uint4*)(h + (rowBase + r) * HID + c * 4);
        *(uint4*)(sm + SH + r * HSTR + c * 16) = v;
        *(uint4*)(sm + SG + r * HSTR + c * 16) = v;
        uint2 p;
        p.x = (u32)f2bf(__uint_as_float(v.x)) |
              ((u32)f2bf(__uint_as_float(v.y)) << 16);
        p.y = (u32)f2bf(__uint_as_float(v.z)) |
              ((u32)f2bf(__uint_as_float(v.w)) << 16);
        *(uint2*)(sm + SY + r * YSTR + c * 8) = p;
    }
    __syncthreads();

    for (int s = 0; s < 4; s++) {
        const float ts = t0 + ((s == 0) ? 0.f : (s == 3) ? DT : 0.5f * DT);
        v8f acc[4];
        // ---- GEMM1: z = tanh(y @ W1 + ts*w1row + b1) ----
#pragma unroll
        for (int t = 0; t < 4; t++) acc[t] = splat8(rb1[t] + ts * rwt[t]);
        gemm4<8, false>(sm + SY, YSTR, sm + SW1, WSTRIDE, l16, hl, ncb, acc);
#pragma unroll
        for (int t = 0; t < 4; t++) {
            const int ncol = ncb + t * 16;
#pragma unroll
            for (int r = 0; r < 8; r++) {
                int m = r + hl * 8;
                *(u16*)(sm + SZ + m * YSTR + ncol * 2) = f2bf(tanhf(acc[t][r]));
            }
        }
        __syncthreads();
        // ---- GEMM2: dh = z @ W2 + b2; fused RK4 epilogue ----
        const float gw = (DT / 6.f) * ((s == 0 || s == 3) ? 1.f : 2.f);
        const float yw = (s <= 1) ? 0.5f * DT : (s == 2) ? DT : 0.f;
#pragma unroll
        for (int t = 0; t < 4; t++) acc[t] = splat8(rb2[t]);
        gemm4<8, false>(sm + SZ, YSTR, sm + SW2, WSTRIDE, l16, hl, ncb, acc);
#pragma unroll
        for (int t = 0; t < 4; t++) {
            const int ncol = ncb + t * 16;
#pragma unroll
            for (int r = 0; r < 8; r++) {
                int m = r + hl * 8;
                float dh = acc[t][r];
                float* gp = (float*)(sm + SG + m * HSTR + ncol * 4);
                *gp += gw * dh;
                if (s < 3) {  // next stage input y = h + c_s*dh
                    float hv = *(const float*)(sm + SH + m * HSTR + ncol * 4);
                    *(u16*)(sm + SY + m * YSTR + ncol * 2) =
                        f2bf(hv + yw * dh);
                }
            }
        }
        __syncthreads();
    }
    // h <- sG (= h_next)
    for (int idx = tid; idx < 16 * 64; idx += 128) {
        int r = idx >> 6, c = idx & 63;
        *(uint4*)(h + (rowBase + r) * HID + c * 4) =
            *(const uint4*)(sm + SG + r * HSTR + c * 16);
    }
}

// ---------------------------------------------------------------------------
// out = hT @ W_out + b_out   ([16384,256]x[256,18]) — memory-bound, plain f32
__global__ void out_kernel(const float* __restrict__ h,
                           const float* __restrict__ Wout,
                           const float* __restrict__ bout,
                           float* __restrict__ out) {
    int idx = blockIdx.x * blockDim.x + threadIdx.x;
    if (idx >= BATCH * OUT_DIM) return;
    int b = idx / OUT_DIM, o = idx % OUT_DIM;
    const float* hr = h + (size_t)b * HID;
    float acc = bout[o];
#pragma unroll 4
    for (int k = 0; k < HID; k++) acc += hr[k] * Wout[k * OUT_DIM + o];
    out[idx] = acc;
}

// ---------------------------------------------------------------------------
extern "C" void kernel_launch(void* const* d_in, const int* in_sizes, int n_in,
                              void* d_out, int out_size, void* d_ws,
                              size_t ws_size, hipStream_t stream) {
    const float* x       = (const float*)d_in[0];
    const float* W_state = (const float*)d_in[1];
    const float* b_state = (const float*)d_in[2];
    const float* W_dyn1  = (const float*)d_in[3];
    const float* b_dyn1  = (const float*)d_in[4];
    const float* W_dyn2  = (const float*)d_in[5];
    const float* b_dyn2  = (const float*)d_in[6];
    const float* W_out   = (const float*)d_in[7];
    const float* b_out   = (const float*)d_in[8];
    float* out = (float*)d_out;

    char* ws = (char*)d_ws;
    float* h  = (float*)(ws + WS_H);
    u16* xbf  = (u16*)(ws + WS_XBF);
    u16* Wst  = (u16*)(ws + WS_WST);
    u16* W1t  = (u16*)(ws + WS_W1T);
    u16* W2t  = (u16*)(ws + WS_W2T);

    cvt_x_kernel<<<4096, 256, 0, stream>>>(x, xbf, BATCH * IN_DIM);
    prep_w_kernel<<<256, 512, 0, stream>>>(W_state, W_dyn1, W_dyn2, Wst, W1t,
                                           W2t);
    h0_kernel<<<BATCH / 16, 128, SMEM_H0, stream>>>(xbf, Wst, b_state, h);
    for (int step = 0; step < NSTEPS; step++) {
        float t0 = (float)step * DT;
        rk4_step_kernel<<<BATCH / 16, 128, SMEM_RK4, stream>>>(
            h, W1t, W2t, W_dyn1, b_dyn1, b_dyn2, t0);
    }
    out_kernel<<<(BATCH * OUT_DIM + 255) / 256, 256, 0, stream>>>(h, W_out,
                                                                  b_out, out);
}
